// MICRO_9380208574580
// MI455X (gfx1250) — matrix operand order repaired and checked by reading, not compiled, not run on hardware
//
#include <hip/hip_runtime.h>
#include <math.h>

#define N_USERS   16384
#define N_ITEMS   8192
#define D_EMB     64
#define IMG_D     4096
#define TXT_D     384
#define TOPK      10
#define NNZ       786432
#define N_TOTAL   (N_USERS + N_ITEMS)
#define KNN_SPLIT 8
#define LAMBDA    0.9f
#define LRELU     0.01f

typedef __attribute__((ext_vector_type(16))) _Float16 v16h;
typedef __attribute__((ext_vector_type(8)))  _Float16 v8h;
typedef __attribute__((ext_vector_type(8)))  float    v8f;

union H16 { v16h v; v8h h[2]; };

// A/B fragment bundle for one 16x(4*16) output strip, K-chunk of 32.
struct Frag {
  H16 a;
  H16 b[4];
};

// ---------------------------------------------------------------------------
// Row L2-normalize (eps-guarded) + convert to f16; also saves the row norm so
// x can be reconstructed as norm * z for the WMMA projection kernel.
// ---------------------------------------------------------------------------
__global__ __launch_bounds__(256)
void l2norm_f16_kernel(const float* __restrict__ x, _Float16* __restrict__ z,
                       float* __restrict__ norms, int D) {
  const int row = blockIdx.x;
  const int tid = threadIdx.x;
  __shared__ float red[256];
  float ss = 0.f;
  for (int j = tid; j < D; j += 256) {
    float v = x[(size_t)row * D + j];
    ss += v * v;
  }
  red[tid] = ss;
  __syncthreads();
  for (int s = 128; s > 0; s >>= 1) {
    if (tid < s) red[tid] += red[tid + s];
    __syncthreads();
  }
  float nrm  = fmaxf(sqrtf(red[0]), 1e-12f);
  float rinv = 1.0f / nrm;
  if (tid == 0) norms[row] = nrm;
  for (int j = tid; j < D; j += 256)
    z[(size_t)row * D + j] = (_Float16)(x[(size_t)row * D + j] * rinv);
}

// ---------------------------------------------------------------------------
// Fused sim-tile GEMM (WMMA f16 -> f32) + per-row running top-10, software-
// pipelined: next K-chunk fragments are loaded while current chunk's 4 WMMAs
// execute, so s_wait_loadcnt overlaps with matrix math instead of stalling.
// One wave owns a 16-row tile and one column strip.
// ---------------------------------------------------------------------------
__global__ __launch_bounds__(32)
void knn_topk_kernel(const _Float16* __restrict__ z, int N, int K, int colsPerStrip,
                     float* __restrict__ part_v, int* __restrict__ part_i) {
  const int lane   = threadIdx.x;
  const int row0   = blockIdx.x * 16;
  const int strip  = blockIdx.y;
  const int cs     = strip * colsPerStrip;
  const int ce     = cs + colsPerStrip;
  const int mrow   = lane & 15;
  const int hiHalf = (lane >> 4) & 1;

  __shared__ float tile[4][16][17];

  float tv[TOPK];
  int   ti[TOPK];
#pragma unroll
  for (int j = 0; j < TOPK; ++j) { tv[j] = -1e30f; ti[j] = 0; }

  const _Float16* arow = z + (size_t)(row0 + mrow) * K;
  const int aoff0 = hiHalf ? 8 : 0;      // ISA 7.12.2: A f16 16x32 layout
  const int aoff1 = hiHalf ? 24 : 16;
  const int boff  = hiHalf ? 8 : 0;      // B f16 32x16: same K order as A

  for (int c0 = cs; c0 < ce; c0 += 64) {   // 4 column tiles per pass
    v8f acc[4] = {};
    const _Float16* brow[4];
#pragma unroll
    for (int t = 0; t < 4; ++t)
      brow[t] = z + (size_t)(c0 + t * 16 + mrow) * K;

    // ---- software-pipelined K loop ----
    Frag cur, nxt;
    cur.a.h[0] = *(const v8h*)(arow + aoff0);
    cur.a.h[1] = *(const v8h*)(arow + aoff1);
#pragma unroll
    for (int t = 0; t < 4; ++t) {
      const _Float16* bp = brow[t] + boff;
      cur.b[t].h[0] = *(const v8h*)(bp);
      cur.b[t].h[1] = *(const v8h*)(bp + 16);
    }
#pragma unroll 2
    for (int kb = 32; kb < K; kb += 32) {
      __builtin_prefetch(arow + kb + 32, 0, 3);
      nxt.a.h[0] = *(const v8h*)(arow + kb + aoff0);
      nxt.a.h[1] = *(const v8h*)(arow + kb + aoff1);
#pragma unroll
      for (int t = 0; t < 4; ++t) {
        const _Float16* bp = brow[t] + kb + boff;
        nxt.b[t].h[0] = *(const v8h*)(bp);
        nxt.b[t].h[1] = *(const v8h*)(bp + 16);
      }
#pragma unroll
      for (int t = 0; t < 4; ++t)
        acc[t] = __builtin_amdgcn_wmma_f32_16x16x32_f16(
            false, cur.a.v, false, cur.b[t].v, (short)0, acc[t], false, false);
      cur = nxt;
    }
#pragma unroll
    for (int t = 0; t < 4; ++t)
      acc[t] = __builtin_amdgcn_wmma_f32_16x16x32_f16(
          false, cur.a.v, false, cur.b[t].v, (short)0, acc[t], false, false);

    // Spill C tiles (acc[v] = elem(M=v+8*hiHalf, N=lane&15)) to LDS,
    // then row-owner lanes run the pruned top-10 insertion.
    __syncthreads();
#pragma unroll
    for (int t = 0; t < 4; ++t) {
#pragma unroll
      for (int v = 0; v < 8; ++v)
        tile[t][v + 8 * hiHalf][lane & 15] = acc[t][v];
    }
    __syncthreads();

    if (lane < 16) {
      for (int t = 0; t < 4; ++t) {
        const int cbase = c0 + t * 16;
        for (int j = 0; j < 16; ++j) {
          float val = tile[t][lane][j];
          if (val > tv[TOPK - 1]) {
            float cv = val;
            int   ci = cbase + j;
#pragma unroll
            for (int p = 0; p < TOPK; ++p) {
              if (cv > tv[p]) {
                float tf = tv[p]; tv[p] = cv; cv = tf;
                int   tn = ti[p]; ti[p] = ci; ci = tn;
              }
            }
          }
        }
      }
    }
    __syncthreads();
  }

  if (lane < 16) {
    const size_t base = ((size_t)(row0 + lane) * KNN_SPLIT + strip) * TOPK;
#pragma unroll
    for (int j = 0; j < TOPK; ++j) {
      part_v[base + j] = tv[j];
      part_i[base + j] = ti[j];
    }
  }
}

// ---------------------------------------------------------------------------
// WMMA projection: y[row,col] = norms[row] * (z_row . wt_col) + bias[col]
// wt is W pre-transposed to [64 x K] f16, so B fragments stream contiguously
// along K exactly like the similarity kernel. Same software pipeline.
// ---------------------------------------------------------------------------
__global__ __launch_bounds__(32)
void wmma_linear_kernel(const _Float16* __restrict__ z, const _Float16* __restrict__ wt,
                        const float* __restrict__ norms, const float* __restrict__ bias,
                        float* __restrict__ y, int K) {
  const int lane   = threadIdx.x;
  const int row0   = blockIdx.x * 16;
  const int mrow   = lane & 15;
  const int hiHalf = (lane >> 4) & 1;

  const _Float16* arow = z + (size_t)(row0 + mrow) * K;
  const int aoff0 = hiHalf ? 8 : 0;
  const int aoff1 = hiHalf ? 24 : 16;
  const int boff  = hiHalf ? 8 : 0;

  v8f acc[4] = {};
  const _Float16* brow[4];
#pragma unroll
  for (int t = 0; t < 4; ++t)
    brow[t] = wt + (size_t)(t * 16 + mrow) * K;

  Frag cur, nxt;
  cur.a.h[0] = *(const v8h*)(arow + aoff0);
  cur.a.h[1] = *(const v8h*)(arow + aoff1);
#pragma unroll
  for (int t = 0; t < 4; ++t) {
    const _Float16* bp = brow[t] + boff;
    cur.b[t].h[0] = *(const v8h*)(bp);
    cur.b[t].h[1] = *(const v8h*)(bp + 16);
  }
#pragma unroll 2
  for (int kb = 32; kb < K; kb += 32) {
    nxt.a.h[0] = *(const v8h*)(arow + kb + aoff0);
    nxt.a.h[1] = *(const v8h*)(arow + kb + aoff1);
#pragma unroll
    for (int t = 0; t < 4; ++t) {
      const _Float16* bp = brow[t] + kb + boff;
      nxt.b[t].h[0] = *(const v8h*)(bp);
      nxt.b[t].h[1] = *(const v8h*)(bp + 16);
    }
#pragma unroll
    for (int t = 0; t < 4; ++t)
      acc[t] = __builtin_amdgcn_wmma_f32_16x16x32_f16(
          false, cur.a.v, false, cur.b[t].v, (short)0, acc[t], false, false);
    cur = nxt;
  }
#pragma unroll
  for (int t = 0; t < 4; ++t)
    acc[t] = __builtin_amdgcn_wmma_f32_16x16x32_f16(
        false, cur.a.v, false, cur.b[t].v, (short)0, acc[t], false, false);

  const int ncol = lane & 15;
#pragma unroll
  for (int t = 0; t < 4; ++t) {
    const int col = t * 16 + ncol;
    const float bc = bias[col];
#pragma unroll
    for (int v = 0; v < 8; ++v) {
      const int m = v + 8 * hiHalf;
      y[(size_t)(row0 + m) * D_EMB + col] =
          norms[row0 + m] * acc[t][v] + bc;
    }
  }
}

// W (K x 64, fp32) -> Wt (64 x K, f16)
__global__ void convert_wt_kernel(const float* __restrict__ w, _Float16* __restrict__ wt,
                                  int K) {
  int idx = blockIdx.x * blockDim.x + threadIdx.x;
  if (idx >= K * D_EMB) return;
  int k = idx / D_EMB, d = idx % D_EMB;   // coalesced read of w
  wt[(size_t)d * K + k] = (_Float16)w[idx];
}

// Merge KNN_SPLIT partial top-10 lists per row into the final top-10.
__global__ void knn_merge_kernel(const float* __restrict__ pv, const int* __restrict__ pi,
                                 float* __restrict__ out_v, int* __restrict__ out_i, int N) {
  int n = blockIdx.x * blockDim.x + threadIdx.x;
  if (n >= N) return;
  float tv[TOPK];
  int   ti[TOPK];
#pragma unroll
  for (int j = 0; j < TOPK; ++j) { tv[j] = -1e30f; ti[j] = 0; }
  const size_t base = (size_t)n * KNN_SPLIT * TOPK;
  for (int s = 0; s < KNN_SPLIT * TOPK; ++s) {
    float cv = pv[base + s];
    int   ci = pi[base + s];
    if (cv > tv[TOPK - 1]) {
#pragma unroll
      for (int p = 0; p < TOPK; ++p) {
        if (cv > tv[p]) {
          float tf = tv[p]; tv[p] = cv; cv = tf;
          int   tn = ti[p]; ti[p] = ci; ci = tn;
        }
      }
    }
  }
#pragma unroll
  for (int j = 0; j < TOPK; ++j) {
    out_v[(size_t)n * TOPK + j] = tv[j];
    out_i[(size_t)n * TOPK + j] = ti[j];
  }
}

// d_inv_sqrt = where(rowsum > 0, rowsum, 1)^(-1/2)
__global__ void knn_dinv_kernel(const float* __restrict__ v, float* __restrict__ dinv, int N) {
  int n = blockIdx.x * blockDim.x + threadIdx.x;
  if (n >= N) return;
  float s = 0.f;
#pragma unroll
  for (int j = 0; j < TOPK; ++j) s += v[(size_t)n * TOPK + j];
  float r = (s > 0.f) ? s : 1.0f;
  dinv[n] = 1.0f / sqrtf(r);
}

// vals[n,k] *= dinv[n] * dinv[inds[n,k]]
__global__ void knn_scale_kernel(float* __restrict__ v, const int* __restrict__ inds,
                                 const float* __restrict__ dinv, int N) {
  int idx = blockIdx.x * blockDim.x + threadIdx.x;
  if (idx >= N * TOPK) return;
  int n = idx / TOPK;
  v[idx] = v[idx] * dinv[n] * dinv[inds[idx]];
}

// out = (1-LAMBDA)*knn_mm(new) + LAMBDA*knn_mm(old), e = item_emb
__global__ __launch_bounds__(64)
void knn_combine_kernel(const float* __restrict__ vn, const int* __restrict__ in_,
                        const float* __restrict__ vo, const int* __restrict__ io_,
                        const float* __restrict__ e, float* __restrict__ out) {
  const int n = blockIdx.x, d = threadIdx.x;
  float sn = 0.f, so = 0.f;
#pragma unroll
  for (int k = 0; k < TOPK; ++k) {
    sn = fmaf(vn[(size_t)n * TOPK + k], e[(size_t)in_[(size_t)n * TOPK + k] * D_EMB + d], sn);
    so = fmaf(vo[(size_t)n * TOPK + k], e[(size_t)io_[(size_t)n * TOPK + k] * D_EMB + d], so);
  }
  out[(size_t)n * D_EMB + d] = (1.0f - LAMBDA) * sn + LAMBDA * so;
}

// Attention gate + fused modality mix:  h = softmax([q(img_e), q(txt_e)]) . [img_e, txt_e]
__global__ __launch_bounds__(64)
void att_kernel(const float* __restrict__ img_e, const float* __restrict__ txt_e,
                const float* __restrict__ q_w1, const float* __restrict__ q_b1,
                const float* __restrict__ q_w2, float* __restrict__ h) {
  const int n = blockIdx.x, d = threadIdx.x;
  __shared__ float xi[64], xt[64], red[64];
  xi[d] = img_e[(size_t)n * D_EMB + d];
  xt[d] = txt_e[(size_t)n * D_EMB + d];
  __syncthreads();
  float si = q_b1[d], st = q_b1[d];
  for (int j = 0; j < 64; ++j) {
    float w = q_w1[(size_t)j * D_EMB + d];
    si = fmaf(xi[j], w, si);
    st = fmaf(xt[j], w, st);
  }
  float qi = tanhf(si) * q_w2[d];
  float qt = tanhf(st) * q_w2[d];

  red[d] = qi; __syncthreads();
  for (int s = 32; s > 0; s >>= 1) { if (d < s) red[d] += red[d + s]; __syncthreads(); }
  float ai = red[0]; __syncthreads();
  red[d] = qt; __syncthreads();
  for (int s = 32; s > 0; s >>= 1) { if (d < s) red[d] += red[d + s]; __syncthreads(); }
  float at = red[0]; __syncthreads();

  float m  = fmaxf(ai, at);
  float e0 = expf(ai - m), e1 = expf(at - m);
  float w0 = e0 / (e0 + e1), w1 = e1 / (e0 + e1);
  h[(size_t)n * D_EMB + d] = w0 * xi[d] + w1 * xt[d];
}

__global__ void concat_kernel(const float* __restrict__ u, const float* __restrict__ it,
                              float* __restrict__ ego, float* __restrict__ acc) {
  int idx = blockIdx.x * blockDim.x + threadIdx.x;
  if (idx >= N_TOTAL * D_EMB) return;
  float v = (idx < N_USERS * D_EMB) ? u[idx] : it[idx - N_USERS * D_EMB];
  ego[idx] = v;
  acc[idx] = v;
}

__global__ void zero_kernel(float* __restrict__ p, int n) {
  int idx = blockIdx.x * blockDim.x + threadIdx.x;
  if (idx < n) p[idx] = 0.f;
}

// side += segment_sum(adj_vals * ego[cols]) via f32 global atomics
__global__ void scatter_kernel(const int* __restrict__ rows, const int* __restrict__ cols,
                               const float* __restrict__ vals, const float* __restrict__ ego,
                               float* __restrict__ side, int nnz) {
  int idx = blockIdx.x * blockDim.x + threadIdx.x;
  if (idx >= nnz * D_EMB) return;
  int e = idx >> 6, d = idx & 63;
  atomicAdd(&side[(size_t)rows[e] * D_EMB + d],
            vals[e] * ego[(size_t)cols[e] * D_EMB + d]);
}

// ego = lrelu(side@gcW+gcB) + lrelu((ego*side)@biW+biB);  acc += l2norm(ego)
__global__ __launch_bounds__(64)
void gcn_dense_kernel(const float* __restrict__ side, float* __restrict__ ego,
                      float* __restrict__ acc,
                      const float* __restrict__ gcw, const float* __restrict__ gcb,
                      const float* __restrict__ biw, const float* __restrict__ bib) {
  const int row = blockIdx.x, d = threadIdx.x;
  __shared__ float ss[64], eg[64], red[64];
  ss[d] = side[(size_t)row * D_EMB + d];
  eg[d] = ego[(size_t)row * D_EMB + d];
  __syncthreads();
  float s = gcb[d], b = bib[d];
  for (int j = 0; j < 64; ++j) {
    float sv = ss[j];
    s = fmaf(sv, gcw[(size_t)j * D_EMB + d], s);
    b = fmaf(sv * eg[j], biw[(size_t)j * D_EMB + d], b);
  }
  s = (s > 0.f) ? s : LRELU * s;
  b = (b > 0.f) ? b : LRELU * b;
  float ne = s + b;
  ego[(size_t)row * D_EMB + d] = ne;
  red[d] = ne * ne; __syncthreads();
  for (int st = 32; st > 0; st >>= 1) { if (d < st) red[d] += red[d + st]; __syncthreads(); }
  float rinv = 1.0f / fmaxf(sqrtf(red[0]), 1e-12f);
  acc[(size_t)row * D_EMB + d] += ne * rinv;
}

// u_g = acc/3 (users);  i_g = acc/3 + l2norm(h) (items)
__global__ __launch_bounds__(64)
void final_kernel(const float* __restrict__ acc, const float* __restrict__ h,
                  float* __restrict__ u_g, float* __restrict__ i_g) {
  const int row = blockIdx.x, d = threadIdx.x;
  if (row < N_USERS) {
    u_g[(size_t)row * D_EMB + d] = acc[(size_t)row * D_EMB + d] * (1.0f / 3.0f);
  } else {
    const int n = row - N_USERS;
    __shared__ float red[64];
    float hv = h[(size_t)n * D_EMB + d];
    red[d] = hv * hv; __syncthreads();
    for (int s = 32; s > 0; s >>= 1) { if (d < s) red[d] += red[d + s]; __syncthreads(); }
    float rinv = 1.0f / fmaxf(sqrtf(red[0]), 1e-12f);
    i_g[(size_t)n * D_EMB + d] = acc[(size_t)row * D_EMB + d] * (1.0f / 3.0f) + hv * rinv;
  }
}

// ---------------------------------------------------------------------------
extern "C" void kernel_launch(void* const* d_in, const int* in_sizes, int n_in,
                              void* d_out, int out_size, void* d_ws, size_t ws_size,
                              hipStream_t stream) {
  (void)in_sizes; (void)n_in; (void)out_size; (void)ws_size;

  const int*   adj_rows    = (const int*)d_in[0];
  const int*   adj_cols    = (const int*)d_in[1];
  const float* adj_vals    = (const float*)d_in[2];
  const float* image_feats = (const float*)d_in[3];
  const float* text_feats  = (const float*)d_in[4];
  const float* user_emb    = (const float*)d_in[5];
  const float* item_emb    = (const float*)d_in[6];
  const float* img_w       = (const float*)d_in[7];
  const float* img_b       = (const float*)d_in[8];
  const float* txt_w       = (const float*)d_in[9];
  const float* txt_b       = (const float*)d_in[10];
  const float* q_w1        = (const float*)d_in[11];
  const float* q_b1        = (const float*)d_in[12];
  const float* q_w2        = (const float*)d_in[13];
  const float* gc_w        = (const float*)d_in[14];
  const float* gc_b        = (const float*)d_in[15];
  const float* bi_w        = (const float*)d_in[16];
  const float* bi_b        = (const float*)d_in[17];

  // d_out layout: [u_g | i_g | img_e | txt_e | h], all fp32
  float* out   = (float*)d_out;
  float* u_g   = out;
  float* i_g   = u_g + (size_t)N_USERS * D_EMB;
  float* img_e = i_g + (size_t)N_ITEMS * D_EMB;
  float* txt_e = img_e + (size_t)N_ITEMS * D_EMB;
  float* h_out = txt_e + (size_t)N_ITEMS * D_EMB;

  // Workspace carve (256B aligned)
  char* wp = (char*)d_ws;
  auto carve = [&](size_t bytes) -> void* {
    void* p = (void*)wp;
    wp += (bytes + 255) & ~(size_t)255;
    return p;
  };
  _Float16* z_img   = (_Float16*)carve((size_t)N_ITEMS * IMG_D * sizeof(_Float16));
  _Float16* z_txt   = (_Float16*)carve((size_t)N_ITEMS * TXT_D * sizeof(_Float16));
  float*    img_f   = (float*)carve((size_t)N_ITEMS * D_EMB * sizeof(float));
  float*    txt_f   = (float*)carve((size_t)N_ITEMS * D_EMB * sizeof(float));
  _Float16* zn_img  = (_Float16*)carve((size_t)N_ITEMS * D_EMB * sizeof(_Float16));
  _Float16* zn_txt  = (_Float16*)carve((size_t)N_ITEMS * D_EMB * sizeof(_Float16));
  _Float16* wt_img  = (_Float16*)carve((size_t)IMG_D * D_EMB * sizeof(_Float16));
  _Float16* wt_txt  = (_Float16*)carve((size_t)TXT_D * D_EMB * sizeof(_Float16));
  float*    nrm_img = (float*)carve((size_t)N_ITEMS * sizeof(float));
  float*    nrm_txt = (float*)carve((size_t)N_ITEMS * sizeof(float));
  float*    nrm_tmp = (float*)carve((size_t)N_ITEMS * sizeof(float));
  float*    part_v  = (float*)carve((size_t)N_ITEMS * KNN_SPLIT * TOPK * sizeof(float));
  int*      part_i  = (int*)carve((size_t)N_ITEMS * KNN_SPLIT * TOPK * sizeof(int));
  float*    io_v    = (float*)carve((size_t)N_ITEMS * TOPK * sizeof(float));
  int*      io_i    = (int*)carve((size_t)N_ITEMS * TOPK * sizeof(int));
  float*    to_v    = (float*)carve((size_t)N_ITEMS * TOPK * sizeof(float));
  int*      to_i    = (int*)carve((size_t)N_ITEMS * TOPK * sizeof(int));
  float*    in_v    = (float*)carve((size_t)N_ITEMS * TOPK * sizeof(float));
  int*      in_i    = (int*)carve((size_t)N_ITEMS * TOPK * sizeof(int));
  float*    tn_v    = (float*)carve((size_t)N_ITEMS * TOPK * sizeof(float));
  int*      tn_i    = (int*)carve((size_t)N_ITEMS * TOPK * sizeof(int));
  float*    dinv    = (float*)carve((size_t)N_ITEMS * sizeof(float));
  float*    ego     = (float*)carve((size_t)N_TOTAL * D_EMB * sizeof(float));
  float*    side    = (float*)carve((size_t)N_TOTAL * D_EMB * sizeof(float));
  float*    accb    = (float*)carve((size_t)N_TOTAL * D_EMB * sizeof(float));

  auto run_knn = [&](const _Float16* z, int K, float* vals, int* inds) {
    dim3 grid(N_ITEMS / 16, KNN_SPLIT);
    knn_topk_kernel<<<grid, 32, 0, stream>>>(z, N_ITEMS, K, N_ITEMS / KNN_SPLIT,
                                             part_v, part_i);
    knn_merge_kernel<<<(N_ITEMS + 127) / 128, 128, 0, stream>>>(part_v, part_i, vals,
                                                                inds, N_ITEMS);
    knn_dinv_kernel<<<(N_ITEMS + 255) / 256, 256, 0, stream>>>(vals, dinv, N_ITEMS);
    knn_scale_kernel<<<(N_ITEMS * TOPK + 255) / 256, 256, 0, stream>>>(vals, inds, dinv,
                                                                       N_ITEMS);
  };

  // 1) Raw-feature KNN graphs (the big WMMA GEMMs)
  l2norm_f16_kernel<<<N_ITEMS, 256, 0, stream>>>(image_feats, z_img, nrm_img, IMG_D);
  l2norm_f16_kernel<<<N_ITEMS, 256, 0, stream>>>(text_feats, z_txt, nrm_txt, TXT_D);
  run_knn(z_img, IMG_D, io_v, io_i);
  run_knn(z_txt, TXT_D, to_v, to_i);

  // 2) Projected-feature KNN graphs: x@W reconstructed as norm*(z@Wt) via WMMA
  convert_wt_kernel<<<(IMG_D * D_EMB + 255) / 256, 256, 0, stream>>>(img_w, wt_img, IMG_D);
  convert_wt_kernel<<<(TXT_D * D_EMB + 255) / 256, 256, 0, stream>>>(txt_w, wt_txt, TXT_D);
  wmma_linear_kernel<<<N_ITEMS / 16, 32, 0, stream>>>(z_img, wt_img, nrm_img, img_b,
                                                      img_f, IMG_D);
  wmma_linear_kernel<<<N_ITEMS / 16, 32, 0, stream>>>(z_txt, wt_txt, nrm_txt, txt_b,
                                                      txt_f, TXT_D);
  l2norm_f16_kernel<<<N_ITEMS, 256, 0, stream>>>(img_f, zn_img, nrm_tmp, D_EMB);
  l2norm_f16_kernel<<<N_ITEMS, 256, 0, stream>>>(txt_f, zn_txt, nrm_tmp, D_EMB);
  run_knn(zn_img, D_EMB, in_v, in_i);
  run_knn(zn_txt, D_EMB, tn_v, tn_i);

  // 3) Modality embeddings + attention fusion
  knn_combine_kernel<<<N_ITEMS, 64, 0, stream>>>(in_v, in_i, io_v, io_i, item_emb, img_e);
  knn_combine_kernel<<<N_ITEMS, 64, 0, stream>>>(tn_v, tn_i, to_v, to_i, item_emb, txt_e);
  att_kernel<<<N_ITEMS, 64, 0, stream>>>(img_e, txt_e, q_w1, q_b1, q_w2, h_out);

  // 4) 2-layer graph conv with running mean of l2-normalized layers
  concat_kernel<<<(N_TOTAL * D_EMB + 255) / 256, 256, 0, stream>>>(user_emb, item_emb,
                                                                   ego, accb);
  for (int layer = 0; layer < 2; ++layer) {
    zero_kernel<<<(N_TOTAL * D_EMB + 255) / 256, 256, 0, stream>>>(side, N_TOTAL * D_EMB);
    scatter_kernel<<<(NNZ * D_EMB + 255) / 256, 256, 0, stream>>>(adj_rows, adj_cols,
                                                                  adj_vals, ego, side, NNZ);
    gcn_dense_kernel<<<N_TOTAL, 64, 0, stream>>>(
        side, ego, accb,
        gc_w + (size_t)layer * D_EMB * D_EMB, gc_b + (size_t)layer * D_EMB,
        bi_w + (size_t)layer * D_EMB * D_EMB, bi_b + (size_t)layer * D_EMB);
  }

  // 5) Final outputs
  final_kernel<<<N_TOTAL, 64, 0, stream>>>(accb, h_out, u_g, i_g);
}
